// Encoder_Decoder_30580167147780
// MI455X (gfx1250) — matrix operand order repaired and checked by reading, not compiled, not run on hardware
//
#include <hip/hip_runtime.h>
#include <hip/hip_bf16.h>
#include <cstdint>

#define HDIM 128
#define G3   384
#define LPAD 512

typedef __attribute__((ext_vector_type(16))) _Float16 v16h;
typedef __attribute__((ext_vector_type(8)))  _Float16 v8h;
typedef __attribute__((ext_vector_type(8)))  float    v8f;

// ---------------- elementwise f32 -> f16 convert ----------------
__global__ void k_cvt_f32_f16(const float* __restrict__ src, _Float16* __restrict__ dst, long n) {
    long i = (long)blockIdx.x * blockDim.x + threadIdx.x;
    long stride = (long)gridDim.x * blockDim.x;
    for (; i < n; i += stride) dst[i] = (_Float16)src[i];
}

// ---------------- pack score/box/origin into allf16 cols 128..223 ----------------
__global__ void k_pack_aux(const float* __restrict__ sc, const float* __restrict__ bx,
                           const float* __restrict__ os, _Float16* __restrict__ allf, int N) {
    int i = blockIdx.x * blockDim.x + threadIdx.x;   // over N*32
    if (i >= N * 32) return;
    int row = i >> 5, j = i & 31;
    _Float16* p = allf + (long)row * 224;
    p[128 + j] = (_Float16)sc[i];
    p[160 + j] = (_Float16)bx[i];
    p[192 + j] = (_Float16)os[i];
}

// ---------------- WMMA GEMM:  out[m][n] = relu( sum_k A[m][k]*B[n][k] + bias[n] ) ----------------
// A: row-major [N][K] f16, B: row-major [128][K] f16 (i.e. Wᵀ is applied).
// One 16x16 output tile per wave; 8 waves/block cover 128 output columns.
template<int K, bool OUT_F16>
__global__ __launch_bounds__(256) void k_gemm_relu(const _Float16* __restrict__ A,
                                                   const _Float16* __restrict__ B,
                                                   const float* __restrict__ bias,
                                                   _Float16* __restrict__ out16,
                                                   float* __restrict__ out32,
                                                   int ldo, int N) {
    int wave = threadIdx.x >> 5;
    int lane = threadIdx.x & 31;
    int m0 = blockIdx.x * 16;
    int n0 = wave * 16;

    int M    = lane & 15;
    int hi8  = (lane & 16) ? 8 : 0;    // unused; A and B both use hi16
    int hi16 = (lane & 16) ? 16 : 0;   // A and B K-offset split, contiguous k order (k = 16h + s)
    int nb   = n0 + (lane & 15);       // this lane's output column (B row)

    long mrow = m0 + M; if (mrow > (long)N - 1) mrow = (long)N - 1;
    const _Float16* arow = A + mrow * (long)K;
    const _Float16* brow = B + (long)nb * K;

    v8f acc = {};
    for (int k0 = 0; k0 < K; k0 += 32) {
        // A 16x32 f16: lane holds row M; halves j -> K=k0+hi16+j (contiguous 16)
        v8h alo = *(const v8h*)(arow + k0 + hi16);
        v8h ahi = *(const v8h*)(arow + k0 + hi16 + 8);
        // B 32x16 f16: lane holds column nb; halves j -> K=k0+hi16+j (contiguous 16)
        v16h bv = *(const v16h*)(brow + k0 + hi16);
        v16h av;
#pragma unroll
        for (int j = 0; j < 8; ++j) { av[j] = alo[j]; av[j + 8] = ahi[j]; }
        acc = __builtin_amdgcn_wmma_f32_16x16x32_f16(false, av, false, bv,
                                                     (short)0, acc, false, false);
    }

    float b = bias[nb];
#pragma unroll
    for (int r = 0; r < 8; ++r) {
        int m = m0 + r + ((lane & 16) ? 8 : 0);   // C/D layout: M = reg + (laneHi?8:0)
        if (m < N) {
            float v = acc[r] + b;
            v = v > 0.f ? v : 0.f;
            if (OUT_F16) out16[(long)m * ldo + nb] = (_Float16)v;
            else         out32[(long)m * ldo + nb] = v;
        }
    }
}

// ---------------- bidirectional GRU scan, one block per (class, direction) ----------------
// LDS: Whh f32 (192KB) + Wih f16 (96KB) + vectors  ~= 293KB of the 320KB WGP LDS.
// Emits per-step scalar  out_W_half . h_t  into sf (fwd) / sr (bwd, scattered to len-1-t).
__global__ __launch_bounds__(384) void k_gru_scan(const float* __restrict__ dec,
                                                  const int* __restrict__ ucl,
                                                  const float* __restrict__ Wih,
                                                  const float* __restrict__ Whh,
                                                  const float* __restrict__ bih,
                                                  const float* __restrict__ bhh,
                                                  const float* __restrict__ outW,
                                                  float* __restrict__ sf,
                                                  float* __restrict__ sr) {
    extern __shared__ char smem[];
    float*    Whh_s = (float*)smem;                                  // 384*128*4 = 196608
    _Float16* Wih_s = (_Float16*)(smem + 196608);                    // 384*128*2 =  98304
    float*    h_s   = (float*)(smem + 294912);                       // 128*4
    float*    x_s   = (float*)(smem + 295424);                       // 128*4
    float*    gx_s  = (float*)(smem + 295936);                       // 384*4
    float*    gh_s  = (float*)(smem + 297472);                       // 384*4
    float*    red   = (float*)(smem + 299008);                       // 128*4  (total 299520)

    int tid = threadIdx.x;
    int cls = blockIdx.x >> 1;
    int dir = blockIdx.x & 1;
    int start = ucl[cls];
    int len   = ucl[cls + 1] - start;

    long wbase = ((long)cls * 2 + dir) * (long)(G3 * HDIM);
    for (int i = tid; i < G3 * HDIM; i += 384) {
        Whh_s[i] = Whh[wbase + i];
        Wih_s[i] = (_Float16)Wih[wbase + i];
    }
    long bbase = ((long)cls * 2 + dir) * G3;
    float bi = bih[bbase + tid];
    float bh = bhh[bbase + tid];
    float wsel = (tid < HDIM) ? outW[dir * HDIM + tid] : 0.f;
    if (tid < HDIM) h_s[tid] = 0.f;
    __syncthreads();

    const float*    wr = Whh_s + tid * HDIM;
    const _Float16* wi = Wih_s + tid * HDIM;

    for (int t = 0; t < len; ++t) {
        int src = (dir == 0) ? (start + t) : (start + len - 1 - t);
        if (tid < HDIM) x_s[tid] = dec[(long)src * HDIM + tid];
        __syncthreads();

        float ax = bi, ah = bh;
#pragma unroll 8
        for (int k = 0; k < HDIM; ++k) {
            ah = fmaf(wr[k], h_s[k], ah);
            ax = fmaf((float)wi[k], x_s[k], ax);
        }
        gx_s[tid] = ax;
        gh_s[tid] = ah;
        __syncthreads();

        if (tid < HDIM) {
            float r = 1.f / (1.f + __expf(-(gx_s[tid] + gh_s[tid])));
            float z = 1.f / (1.f + __expf(-(gx_s[HDIM + tid] + gh_s[HDIM + tid])));
            float n = tanhf(gx_s[2 * HDIM + tid] + r * gh_s[2 * HDIM + tid]);
            float hn = (1.f - z) * n + z * h_s[tid];
            h_s[tid] = hn;
            red[tid] = wsel * hn;
        }
        __syncthreads();

        for (int s = 64; s > 0; s >>= 1) {
            if (tid < s) red[tid] += red[tid + s];
            __syncthreads();
        }
        if (tid == 0) {
            float s = red[0];
            if (dir == 0) sf[cls * LPAD + t] = s;
            else          sr[cls * LPAD + (len - 1 - t)] = s;
        }
        __syncthreads();
    }
}

// ---------------- final combine + sigmoid ----------------
__global__ void k_final(const float* __restrict__ sf, const float* __restrict__ sr,
                        const int* __restrict__ ucl, const float* __restrict__ out_b,
                        float* __restrict__ out) {
    int c = blockIdx.x;
    int start = ucl[c], len = ucl[c + 1] - start;
    float ob = out_b[0];
    for (int t = threadIdx.x; t < len; t += blockDim.x) {
        float s = sf[c * LPAD + t] + sr[c * LPAD + t] + ob;
        out[start + t] = 1.f / (1.f + __expf(-s));
    }
}

extern "C" void kernel_launch(void* const* d_in, const int* in_sizes, int n_in,
                              void* d_out, int out_size, void* d_ws, size_t ws_size,
                              hipStream_t stream) {
    (void)n_in; (void)out_size; (void)ws_size;
    const float* acf   = (const float*)d_in[3];   // all_class_boxes_feature (N,1024)
    const float* acs   = (const float*)d_in[4];   // all_class_boxes_score   (N,32)
    const float* acb   = (const float*)d_in[5];   // all_class_boxes_box     (N,32)
    const float* aco   = (const float*)d_in[6];   // all_class_boxes_origin_score (N,32)
    const int*   ucl   = (const int*)d_in[8];     // unique_class_len (C+1,)
    const float* appW  = (const float*)d_in[9];   // (128,1024)
    const float* appB  = (const float*)d_in[10];  // (128,)
    const float* featW = (const float*)d_in[11];  // (128,224)
    const float* featB = (const float*)d_in[12];  // (128,)
    const float* gWih  = (const float*)d_in[13];  // (C,2,384,128)
    const float* gWhh  = (const float*)d_in[14];  // (C,2,384,128)
    const float* gbih  = (const float*)d_in[15];  // (C,2,384)
    const float* gbhh  = (const float*)d_in[16];  // (C,2,384)
    const float* outW  = (const float*)d_in[17];  // (1,256)
    const float* outB  = (const float*)d_in[18];  // (1,)

    int N = in_sizes[3] / 1024;
    int C = in_sizes[7];

    uintptr_t p = (uintptr_t)d_ws;
    auto carve = [&](size_t bytes) -> void* {
        p = (p + 255) & ~(uintptr_t)255;
        void* r = (void*)p;
        p += bytes;
        return r;
    };
    _Float16* A16    = (_Float16*)carve((size_t)N * 1024 * 2);
    _Float16* allf16 = (_Float16*)carve((size_t)N * 224 * 2);
    _Float16* W16    = (_Float16*)carve((size_t)128 * 1024 * 2);
    _Float16* fW16   = (_Float16*)carve((size_t)128 * 224 * 2);
    float*    dec    = (float*)carve((size_t)N * 128 * 4);
    float*    sf     = (float*)carve((size_t)C * LPAD * 4);
    float*    sr     = (float*)carve((size_t)C * LPAD * 4);

    // 1) f16 conversions
    k_cvt_f32_f16<<<4096, 256, 0, stream>>>(acf, A16, (long)N * 1024);
    k_cvt_f32_f16<<<512, 256, 0, stream>>>(appW, W16, 128L * 1024);
    k_cvt_f32_f16<<<112, 256, 0, stream>>>(featW, fW16, 128L * 224);
    k_pack_aux<<<(N * 32 + 255) / 256, 256, 0, stream>>>(acs, acb, aco, allf16, N);

    // 2) WMMA GEMMs
    int rowTiles = (N + 15) / 16;
    k_gemm_relu<1024, true><<<rowTiles, 256, 0, stream>>>(A16, W16, appB, allf16, nullptr, 224, N);
    k_gemm_relu<224, false><<<rowTiles, 256, 0, stream>>>(allf16, fW16, featB, nullptr, dec, 128, N);

    // 3) GRU scan (one block per class-direction, big dynamic LDS)
    const int SMEM = 299520;
    hipFuncSetAttribute(reinterpret_cast<const void*>(&k_gru_scan),
                        hipFuncAttributeMaxDynamicSharedMemorySize, SMEM);
    k_gru_scan<<<C * 2, 384, SMEM, stream>>>(dec, ucl, gWih, gWhh, gbih, gbhh, outW, sf, sr);

    // 4) combine + sigmoid
    k_final<<<C, 256, 0, stream>>>(sf, sr, ucl, outB, (float*)d_out);
}
